// QNetMLPNAgent_7507602833455
// MI455X (gfx1250) — matrix operand order repaired and checked by reading, not compiled, not run on hardware
//
#include <hip/hip_runtime.h>
#include <hip/hip_bf16.h>

typedef __bf16 bf16;
typedef __attribute__((ext_vector_type(16))) __bf16 v16bf;
typedef __attribute__((ext_vector_type(8)))  __bf16 v8bf;
typedef __attribute__((ext_vector_type(8)))  float  v8f;
typedef __attribute__((ext_vector_type(4)))  unsigned int u32x4;
typedef __attribute__((ext_vector_type(8)))  int i32x8;
typedef __attribute__((ext_vector_type(4)))  int i32x4;

union Frag { v16bf v; v8bf h[2]; };

#define K_TILE 32

#if __has_builtin(__builtin_amdgcn_tensor_load_to_lds) && __has_builtin(__builtin_amdgcn_s_wait_tensorcnt)
#define USE_TDM 1
#if __has_include(<hip/amd_detail/amd_gfx1250_TDM.h>)
#define TDM_6ARG 1
#endif
#else
#define USE_TDM 0
#endif

#if USE_TDM
// ---------------------------------------------------------------------------
// Issue one TDM 2D tile load: tileRows x tileK elements (bf16), row stride
// rowStrideElems, from global tile origin gaddr into LDS at lds_off.
// D# bit layout per CDNA5 ISA 8.3/8.4 (count=1, type=2 "image", data_size=2B).
// ---------------------------------------------------------------------------
__device__ __forceinline__ void tdm_load_2d(unsigned lds_off, const void* gaddr,
                                            unsigned tileK, unsigned tileRows,
                                            unsigned tensorK, unsigned tensorRows,
                                            unsigned rowStrideElems)
{
    unsigned long long ga = (unsigned long long)(size_t)gaddr;
    u32x4 g0;
    g0.x = 1u;                                        // count=1, is_restore=0
    g0.y = lds_off;                                   // lds_addr (bytes)
    g0.z = (unsigned)(ga & 0xFFFFFFFFu);              // global_addr[31:0]
    g0.w = (unsigned)((ga >> 32) & 0x1FFFFFFu)        // global_addr[56:32]
           | (2u << 30);                              // type=2
    i32x8 g1;
    g1[0] = (int)(1u << 16);                          // wg_mask=0, data_size=1 (2B)
    g1[1] = (int)((tensorK & 0xFFFFu) << 16);         // tensor_dim0[15:0]
    g1[2] = (int)((tensorK >> 16) | ((tensorRows & 0xFFFFu) << 16)); // dim0 hi | dim1 lo
    g1[3] = (int)((tensorRows >> 16) | (tileK << 16));// dim1 hi | tile_dim0
    g1[4] = (int)(tileRows & 0xFFFFu);                // tile_dim1 (tile_dim2=0)
    g1[5] = (int)rowStrideElems;                      // tensor_dim0_stride[31:0]
    g1[6] = 0;                                        // stride hi / dim1_stride lo
    g1[7] = 0;
    i32x4 gz = {0, 0, 0, 0};
#if defined(TDM_6ARG)
    i32x8 gz8 = {0, 0, 0, 0, 0, 0, 0, 0};
    __builtin_amdgcn_tensor_load_to_lds(g0, g1, gz, gz, gz8, 0);
#else
    __builtin_amdgcn_tensor_load_to_lds(g0, g1, gz, gz, 0);
#endif
}
__device__ __forceinline__ unsigned lds_off_of(const void* p) {
    return (unsigned)(size_t)p;   // generic LDS ptr low 32 bits = LDS byte offset
}
#endif

// ---------------------------------------------------------------------------
// Prep: build bf16 X = [obs | actions]  (B x 2176)
// ---------------------------------------------------------------------------
__global__ __launch_bounds__(256) void cvt_x(const float* __restrict__ obs,
                                             const float* __restrict__ act,
                                             bf16* __restrict__ Xb) {
    size_t id = (size_t)blockIdx.x * 256 + threadIdx.x;
    int b = (int)(id / 2176);
    int c = (int)(id % 2176);
    float v = (c < 2048) ? obs[(size_t)b * 2048 + c]
                         : act[(size_t)b * 128 + (c - 2048)];
    Xb[id] = (bf16)v;
}

// ---------------------------------------------------------------------------
// Prep: W1 (A,2176,1024) f32 -> W1t (A,1024,2176) bf16, self-action mask folded
// ---------------------------------------------------------------------------
__global__ __launch_bounds__(256) void cvt_w1(const float* __restrict__ W1,
                                              bf16* __restrict__ W1t) {
    size_t id = (size_t)blockIdx.x * 256 + threadIdx.x;
    int a = (int)(id / ((size_t)2176 * 1024));
    size_t rem = id % ((size_t)2176 * 1024);
    int k = (int)(rem / 1024);
    int n = (int)(rem % 1024);
    float v = W1[id];
    if (k >= 2048 && ((k - 2048) >> 4) == a) v = 0.0f;
    W1t[((size_t)a * 1024 + n) * 2176 + k] = (bf16)v;
}

__global__ __launch_bounds__(256) void cvt_w2(const float* __restrict__ W2,
                                              bf16* __restrict__ W2t) {
    size_t id = (size_t)blockIdx.x * 256 + threadIdx.x;
    int a = (int)(id >> 20);
    size_t rem = id & ((1u << 20) - 1);
    int k = (int)(rem >> 10);
    int n = (int)(rem & 1023);
    W2t[((size_t)a * 1024 + n) * 1024 + k] = (bf16)W2[id];
}

__global__ __launch_bounds__(256) void cvt_w3(const float* __restrict__ W3,
                                              bf16* __restrict__ W3t) {
    size_t id = (size_t)blockIdx.x * 256 + threadIdx.x;
    int a = (int)(id / (1024 * 16));
    size_t rem = id % (1024 * 16);
    int k = (int)(rem / 16);
    int n = (int)(rem % 16);
    W3t[((size_t)a * 16 + n) * 1024 + k] = (bf16)W3[id];
}

// ---------------------------------------------------------------------------
// Main GEMM: C[a] = act( A[a] (MxK) * B[a]^T (NxK, N-major) + bias[a] )
// 256 thr = 8 waves. C tile 128x128, wave = 32x64 (2x4 WMMA blocks).
// TDM double-buffered global->LDS (fallback: register-staged copy).
// ---------------------------------------------------------------------------
__global__ __launch_bounds__(256) void gemm_bf16_tile(
    const bf16* __restrict__ Abase, long aStride,
    const bf16* __restrict__ Bbase, long bStride,
    const float* __restrict__ bias, int biasStride,
    bf16* __restrict__ Cbase, long cStride,
    int K, int lda, int ldb, int ldc, int doRelu)
{
    __shared__ bf16 As[2][128][K_TILE];
    __shared__ bf16 Bs[2][128][K_TILE];

    const int ag = blockIdx.z;
    const bf16*  A  = Abase + (size_t)ag * aStride;
    const bf16*  Bw = Bbase + (size_t)ag * bStride;
    const float* bv = bias + (size_t)ag * biasStride;
    bf16*        C  = Cbase + (size_t)ag * cStride;

    const int mTile = blockIdx.x * 128;
    const int nTile = blockIdx.y * 128;

    const int t    = threadIdx.x;
    const int lane = t & 31;
    const int wid  = t >> 5;
    const int l16  = lane & 15;
    const int lh   = lane >> 4;
    const int wm   = (wid & 3) * 32;    // wave row base (2 x 16-row blocks)
    const int wn   = (wid >> 2) * 64;   // wave col base (4 x 16-col blocks)

    v8f acc[2][4] = {};
    const int nk = K / K_TILE;

#if USE_TDM
    if (wid == 0) {
        tdm_load_2d(lds_off_of(&As[0][0][0]), A + (size_t)mTile * lda,
                    K_TILE, 128, (unsigned)lda, 8192u, (unsigned)lda);
        tdm_load_2d(lds_off_of(&Bs[0][0][0]), Bw + (size_t)nTile * ldb,
                    K_TILE, 128, (unsigned)ldb, 1024u, (unsigned)ldb);
        __builtin_amdgcn_s_wait_tensorcnt(0);
    }
    __syncthreads();
#endif

    for (int kt = 0; kt < nk; ++kt) {
        const int buf = kt & 1;
        const int k0 = kt * K_TILE;

#if USE_TDM
        // Overlap: issue next tile's DMA before computing this one.
        if (wid == 0 && kt + 1 < nk) {
            const int k1 = k0 + K_TILE;
            tdm_load_2d(lds_off_of(&As[buf ^ 1][0][0]),
                        A + (size_t)mTile * lda + k1,
                        K_TILE, 128, (unsigned)lda, 8192u, (unsigned)lda);
            tdm_load_2d(lds_off_of(&Bs[buf ^ 1][0][0]),
                        Bw + (size_t)nTile * ldb + k1,
                        K_TILE, 128, (unsigned)ldb, 1024u, (unsigned)ldb);
        }
#else
        // Register-staged copy: 4 x 16B chunks per thread (2 A + 2 B rows).
        {
            const int row0 = t >> 2, row1 = (t + 256) >> 2;
            const int chk  = (t & 3) * 8;
            const uint4 va0 = *reinterpret_cast<const uint4*>(
                A + (size_t)(mTile + row0) * lda + k0 + chk);
            const uint4 va1 = *reinterpret_cast<const uint4*>(
                A + (size_t)(mTile + row1) * lda + k0 + chk);
            const uint4 vb0 = *reinterpret_cast<const uint4*>(
                Bw + (size_t)(nTile + row0) * ldb + k0 + chk);
            const uint4 vb1 = *reinterpret_cast<const uint4*>(
                Bw + (size_t)(nTile + row1) * ldb + k0 + chk);
            if (kt + 1 < nk) {
                __builtin_prefetch(A  + (size_t)(mTile + row0) * lda + k0 + K_TILE + chk, 0, 1);
                __builtin_prefetch(Bw + (size_t)(nTile + row0) * ldb + k0 + K_TILE + chk, 0, 1);
            }
            __syncthreads();
            *reinterpret_cast<uint4*>(&As[buf][row0][chk]) = va0;
            *reinterpret_cast<uint4*>(&As[buf][row1][chk]) = va1;
            *reinterpret_cast<uint4*>(&Bs[buf][row0][chk]) = vb0;
            *reinterpret_cast<uint4*>(&Bs[buf][row1][chk]) = vb1;
            __syncthreads();
        }
#endif

        // Fragments per ISA VGPR maps, then 8 back-to-back WMMAs.
        Frag fa0, fa1;
        fa0.h[0] = *reinterpret_cast<const v8bf*>(&As[buf][wm + l16][lh * 8]);
        fa0.h[1] = *reinterpret_cast<const v8bf*>(&As[buf][wm + l16][lh * 8 + 16]);
        fa1.h[0] = *reinterpret_cast<const v8bf*>(&As[buf][wm + 16 + l16][lh * 8]);
        fa1.h[1] = *reinterpret_cast<const v8bf*>(&As[buf][wm + 16 + l16][lh * 8 + 16]);
        Frag fb[4];
        #pragma unroll
        for (int j = 0; j < 4; ++j) {
            const int col = wn + j * 16 + l16;
            fb[j].h[0] = *reinterpret_cast<const v8bf*>(&Bs[buf][col][lh * 8]);
            fb[j].h[1] = *reinterpret_cast<const v8bf*>(&Bs[buf][col][lh * 8 + 16]);
        }
        #pragma unroll
        for (int j = 0; j < 4; ++j) {
            acc[0][j] = __builtin_amdgcn_wmma_f32_16x16x32_bf16(
                false, fa0.v, false, fb[j].v, (short)0, acc[0][j], false, false);
            acc[1][j] = __builtin_amdgcn_wmma_f32_16x16x32_bf16(
                false, fa1.v, false, fb[j].v, (short)0, acc[1][j], false, false);
        }

#if USE_TDM
        __syncthreads();                               // all waves done reading buf
        if (wid == 0 && kt + 1 < nk)
            __builtin_amdgcn_s_wait_tensorcnt(0);      // next tile landed in LDS
        __syncthreads();                               // publish to all waves
#endif
    }

    // C layout: VGPR r -> row r + 8*lh, col l16
    #pragma unroll
    for (int i = 0; i < 2; ++i) {
        #pragma unroll
        for (int j = 0; j < 4; ++j) {
            const int gcol = nTile + wn + j * 16 + l16;
            const float bb = bv[gcol];
            #pragma unroll
            for (int r = 0; r < 8; ++r) {
                const int grow = mTile + wm + i * 16 + r + lh * 8;
                float v = acc[i][j][r] + bb;
                if (doRelu) v = v > 0.0f ? v : 0.0f;
                C[(size_t)grow * ldc + gcol] = (bf16)v;
            }
        }
    }
}

// ---------------------------------------------------------------------------
// Layer 3: (8192x1024)*(1024x16) + b3 -> f32 out. 8 waves x 16 rows.
// ---------------------------------------------------------------------------
__global__ __launch_bounds__(256) void gemm3_out(
    const bf16* __restrict__ H2, const bf16* __restrict__ W3t,
    const float* __restrict__ b3, float* __restrict__ out)
{
    __shared__ bf16 As[128][K_TILE + 8];
    __shared__ bf16 Bs[16][K_TILE + 8];

    const int ag = blockIdx.y;
    const bf16*  A  = H2 + (size_t)ag * 8192 * 1024;
    const bf16*  Bw = W3t + (size_t)ag * 16 * 1024;
    const float* bv = b3 + ag * 16;

    const int mTile = blockIdx.x * 128;
    const int t    = threadIdx.x;
    const int lane = t & 31;
    const int wid  = t >> 5;
    const int l16  = lane & 15;
    const int lh   = lane >> 4;
    const int m0   = wid * 16;

    v8f acc = {};

    for (int k0 = 0; k0 < 1024; k0 += K_TILE) {
        const int achk = (t & 3) * 8;
        const uint4 va0 = *reinterpret_cast<const uint4*>(
            A + (size_t)(mTile + (t >> 2)) * 1024 + k0 + achk);
        const uint4 va1 = *reinterpret_cast<const uint4*>(
            A + (size_t)(mTile + ((t + 256) >> 2)) * 1024 + k0 + achk);
        uint4 vb = {};
        const bool doB = t < 64;
        if (doB)
            vb = *reinterpret_cast<const uint4*>(
                Bw + (size_t)(t >> 2) * 1024 + k0 + achk);

        __syncthreads();
        *reinterpret_cast<uint4*>(&As[t >> 2][achk])         = va0;
        *reinterpret_cast<uint4*>(&As[(t + 256) >> 2][achk]) = va1;
        if (doB)
            *reinterpret_cast<uint4*>(&Bs[t >> 2][achk]) = vb;
        __syncthreads();

        Frag fa, fb;
        fa.h[0] = *reinterpret_cast<const v8bf*>(&As[m0 + l16][lh * 8]);
        fa.h[1] = *reinterpret_cast<const v8bf*>(&As[m0 + l16][lh * 8 + 16]);
        fb.h[0] = *reinterpret_cast<const v8bf*>(&Bs[l16][lh * 8]);
        fb.h[1] = *reinterpret_cast<const v8bf*>(&Bs[l16][lh * 8 + 16]);
        acc = __builtin_amdgcn_wmma_f32_16x16x32_bf16(
            false, fa.v, false, fb.v, (short)0, acc, false, false);
    }

    const float bb = bv[l16];
    #pragma unroll
    for (int r = 0; r < 8; ++r) {
        const int grow = mTile + m0 + r + lh * 8;
        out[(size_t)grow * 128 + ag * 16 + l16] = acc[r] + bb;
    }
}

// ---------------------------------------------------------------------------
extern "C" void kernel_launch(void* const* d_in, const int* in_sizes, int n_in,
                              void* d_out, int out_size, void* d_ws, size_t ws_size,
                              hipStream_t stream) {
    (void)in_sizes; (void)n_in; (void)out_size; (void)ws_size;
    const float* obs = (const float*)d_in[0];
    const float* act = (const float*)d_in[1];
    const float* W1  = (const float*)d_in[2];
    const float* b1  = (const float*)d_in[3];
    const float* W2  = (const float*)d_in[4];
    const float* b2  = (const float*)d_in[5];
    const float* W3  = (const float*)d_in[6];
    const float* b3  = (const float*)d_in[7];
    float* out = (float*)d_out;

    char* ws = (char*)d_ws;
    size_t off = 0;
    auto take = [&](size_t bytes) -> char* {
        char* p = ws + off;
        off = (off + bytes + 255) & ~(size_t)255;
        return p;
    };
    bf16* Xb  = (bf16*)take((size_t)8192 * 2176 * 2);
    bf16* W1t = (bf16*)take((size_t)8 * 1024 * 2176 * 2);
    bf16* W2t = (bf16*)take((size_t)8 * 1024 * 1024 * 2);
    bf16* W3t = (bf16*)take((size_t)8 * 16 * 1024 * 2);
    bf16* H1  = (bf16*)take((size_t)8 * 8192 * 1024 * 2);
    bf16* H2  = (bf16*)take((size_t)8 * 8192 * 1024 * 2);

    cvt_x <<<69632, 256, 0, stream>>>(obs, act, Xb);
    cvt_w1<<<69632, 256, 0, stream>>>(W1, W1t);
    cvt_w2<<<32768, 256, 0, stream>>>(W2, W2t);
    cvt_w3<<<512,   256, 0, stream>>>(W3, W3t);

    // L1: h1 = relu(X @ W1m[a] + b1[a]);  K = 2176, X shared across agents
    dim3 g1(8192 / 128, 1024 / 128, 8);
    gemm_bf16_tile<<<g1, 256, 0, stream>>>(
        Xb, 0L, W1t, (long)2176 * 1024, b1, 1024,
        H1, (long)8192 * 1024, 2176, 2176, 2176, 1024, 1);

    // L2: h2 = relu(h1 @ W2[a] + b2[a]);  K = 1024
    gemm_bf16_tile<<<g1, 256, 0, stream>>>(
        H1, (long)8192 * 1024, W2t, (long)1024 * 1024, b2, 1024,
        H2, (long)8192 * 1024, 1024, 1024, 1024, 1024, 1);

    // L3: out = h2 @ W3[a] + b3[a]  (f32)
    dim3 g3(8192 / 128, 8);
    gemm3_out<<<g3, 256, 0, stream>>>(H2, W3t, b3, out);
}
